// AFNO2D_1202590843587
// MI455X (gfx1250) — matrix operand order repaired and checked by reading, not compiled, not run on hardware
//
#include <hip/hip_runtime.h>
#include <math.h>

// ---------------------------------------------------------------------------
// AFNO2D on MI455X (gfx1250): all heavy math as WMMA f32_16x16x32_f16 GEMMs.
//   stage1: rfft along W as real GEMM  (K=240 pad 256)
//   stage2: fft along H as complex GEMM (D 240x240 x image 240x128)
//   stage3: block-diagonal complex channel mixing (two 96x96 layers + mask)
//   stage4: inverse fft along H (same kernel, swapped sin matrices)
//   stage5: irfft along W with hermitian fold -> f32 output
// Twiddle matrices built on-device each call (graph capture safe).
// This revision: software-pipelined B fragments, double-buffered LDS staging,
// global_prefetch_b8 latency hiding, coalesced mixing gather.
// ---------------------------------------------------------------------------

typedef __attribute__((ext_vector_type(16))) _Float16 v16h;
typedef __attribute__((ext_vector_type(8)))  _Float16 v8h;
typedef __attribute__((ext_vector_type(8)))  float    v8f;

#define B_      2
#define C_      768
#define H_      240
#define W_      240
#define WF      121
#define WFP     128
#define NB      8
#define BS      96
#define ROWS_W  (B_*C_*H_)        // 368640
#define IMGS    (B_*C_)           // 1536
#define MIXM    (B_*H_*WF)        // 58080
#define PLANE   (B_*C_*H_*WFP)    // 47185920 f16 elems per plane
#define TPI240  0.026179938779914946f  // 2*pi/240

// ----------------------------- small helpers -------------------------------
__device__ __forceinline__ v8f zero8f() {
  v8f z;
#pragma unroll
  for (int e = 0; e < 8; ++e) z[e] = 0.0f;
  return z;
}
__device__ __forceinline__ v8h zero8h() {
  v8h z;
#pragma unroll
  for (int e = 0; e < 8; ++e) z[e] = (_Float16)0.0f;
  return z;
}
__device__ __forceinline__ v16h make_frag(v8h lo, v8h hi) {
  v16h r;
#pragma unroll
  for (int e = 0; e < 8; ++e) { r[e] = lo[e]; r[e + 8] = hi[e]; }
  return r;
}
__device__ __forceinline__ v16h neg16(v16h a) {
  v16h r;
#pragma unroll
  for (int e = 0; e < 16; ++e) r[e] = -a[e];
  return r;
}
// A-fragment: row (M = lane&15) major, K chunks at kk+8*half and kk+16+8*half
__device__ __forceinline__ v16h frag_a(const _Float16* p, int ld, int row,
                                       int kk, int half) {
  const _Float16* q = p + row * ld + kk + 8 * half;
  return make_frag(*(const v8h*)q, *(const v8h*)(q + 16));
}
// B-fragment from N-major (transposed) storage: col = lane&15,
// K chunks at kk+8*half and kk+16+8*half (two 16B chunks)
__device__ __forceinline__ v16h frag_b(const _Float16* p, int ld, int col,
                                       int kk, int half) {
  const _Float16* q = p + col * ld + kk + 8 * half;
  return make_frag(*(const v8h*)q, *(const v8h*)(q + 16));
}
// 8 contiguous f32 -> v8h (f16)
__device__ __forceinline__ v8h cvt8(const float* p) {
  float4 u = ((const float4*)p)[0];
  float4 v = ((const float4*)p)[1];
  v8h r;
  r[0] = (_Float16)u.x; r[1] = (_Float16)u.y; r[2] = (_Float16)u.z; r[3] = (_Float16)u.w;
  r[4] = (_Float16)v.x; r[5] = (_Float16)v.y; r[6] = (_Float16)v.z; r[7] = (_Float16)v.w;
  return r;
}
__device__ __forceinline__ v8f wmma(v16h a, v16h b, v8f c) {
  return __builtin_amdgcn_wmma_f32_16x16x32_f16(false, a, false, b, (short)0,
                                                c, false, false);
}

// ----------------------------- twiddle init --------------------------------
// FwcT/FwsT: [k=128][w=256] transposed forward-W DFT (ortho 1/240 folded in)
__global__ void init_fw(_Float16* FwcT, _Float16* FwsT) {
  int idx = blockIdx.x * 256 + threadIdx.x;
  if (idx >= 128 * 256) return;
  int k = idx >> 8, w = idx & 255;
  float c = 0.f, s = 0.f;
  if (k < WF && w < W_) {
    int m = (k * w) % 240;
    float th = (float)m * TPI240;
    c =  cosf(th) * (1.0f / 240.0f);
    s = -sinf(th) * (1.0f / 240.0f);
  }
  FwcT[idx] = (_Float16)c; FwsT[idx] = (_Float16)s;
}
// P/Sg/Sn: [a=240][b=256] symmetric H-DFT components cos, +sin, -sin
__global__ void init_h(_Float16* P, _Float16* Sg, _Float16* Sn) {
  int idx = blockIdx.x * 256 + threadIdx.x;
  if (idx >= 240 * 256) return;
  int k = idx >> 8, h = idx & 255;
  float c = 0.f, s = 0.f;
  if (h < H_) {
    int m = (k * h) % 240;
    float th = (float)m * TPI240;
    c = cosf(th); s = sinf(th);
  }
  P[idx] = (_Float16)c; Sg[idx] = (_Float16)s; Sn[idx] = (_Float16)(-s);
}
// IwcT/IwsT: [w'=240][k=128] inverse-W with hermitian fold + 1/240
__global__ void init_iw(_Float16* IwcT, _Float16* IwsT) {
  int idx = blockIdx.x * 256 + threadIdx.x;
  if (idx >= 240 * 128) return;
  int w = idx >> 7, k = idx & 127;
  float c = 0.f, s = 0.f;
  if (k < WF) {
    float a = (k == 0 || k == 120) ? 1.0f : 2.0f;
    int m = (k * w) % 240;
    float th = (float)m * TPI240;
    c =  a * cosf(th) * (1.0f / 240.0f);
    s = -a * sinf(th) * (1.0f / 240.0f);
  }
  IwcT[idx] = (_Float16)c; IwsT[idx] = (_Float16)s;
}

// --------------------- stage 1: rfft along W (real GEMM) -------------------
// 128 threads = 4 waves; block tile M=64 (16/wave), N=128, K=240 (8 steps).
// B-fragments are software pipelined (load t+1 while multiplying t).
__global__ __launch_bounds__(128) void rfft_w_kernel(
    const float* __restrict__ x, const _Float16* __restrict__ BcT,
    const _Float16* __restrict__ BsT, _Float16* __restrict__ Yr,
    _Float16* __restrict__ Yi) {
  int wave = threadIdx.x >> 5, lane = threadIdx.x & 31;
  int half = lane >> 4, l15 = lane & 15;
  int rw = blockIdx.x * 64 + wave * 16;
  int ra = rw + l15;
  v8f accr[8], acci[8];
#pragma unroll
  for (int t = 0; t < 8; ++t) { accr[t] = zero8f(); acci[t] = zero8f(); }
#pragma unroll
  for (int ks = 0; ks < 8; ++ks) {
    int kk = ks * 32;
    int c0 = kk + 8 * half, c1 = c0 + 16;
    // prefetch the next A chunk for this row (global_prefetch_b8, counter-free)
    if (c0 + 32 < W_) __builtin_prefetch(x + ra * W_ + c0 + 32, 0, 0);
    v8h lo = (c0 < W_) ? cvt8(x + ra * W_ + c0) : zero8h();
    v8h hi = (c1 < W_) ? cvt8(x + ra * W_ + c1) : zero8h();
    v16h a = make_frag(lo, hi);
    v16h bc = frag_b(BcT, 256, l15, kk, half);
    v16h bs = frag_b(BsT, 256, l15, kk, half);
#pragma unroll
    for (int t = 0; t < 8; ++t) {
      v16h bcn = bc, bsn = bs;
      if (t < 7) {
        int coln = (t + 1) * 16 + l15;
        bcn = frag_b(BcT, 256, coln, kk, half);
        bsn = frag_b(BsT, 256, coln, kk, half);
      }
      accr[t] = wmma(a, bc, accr[t]);
      acci[t] = wmma(a, bs, acci[t]);
      bc = bcn; bs = bsn;
    }
  }
#pragma unroll
  for (int t = 0; t < 8; ++t) {
    int col = t * 16 + l15;
#pragma unroll
    for (int v = 0; v < 8; ++v) {
      int row = rw + v + 8 * half;
      Yr[row * WFP + col] = (_Float16)accr[t][v];
      Yi[row * WFP + col] = (_Float16)acci[t][v];
    }
  }
}

// ---------------- stage 2/4: complex DFT along H (batched GEMM) ------------
// Zr = P*Br + A1*Bi ; Zi = P*Bi + A2*Br.  fwd: A1=+sin, A2=-sin; inv swapped.
// 96 threads = 3 waves; per block M=48 (exact, 5 blocks/image), N=128.
// Double-buffered LDS transpose staging + prefetch of the next k-slab.
__global__ __launch_bounds__(96) void cgemm_h_kernel(
    const _Float16* __restrict__ P, const _Float16* __restrict__ A1,
    const _Float16* __restrict__ A2, const _Float16* __restrict__ Br_,
    const _Float16* __restrict__ Bi_, _Float16* __restrict__ Or,
    _Float16* __restrict__ Oi) {
  __shared__ __align__(16) _Float16 BT[2][2][128 * 32];  // [plane][buf]
  int tid = threadIdx.x;
  int wave = tid >> 5, lane = tid & 31, half = lane >> 4, l15 = lane & 15;
  int img = blockIdx.y;
  int rw = blockIdx.x * 48 + wave * 16;
  int ra = rw + l15;
  const _Float16* Bsr = Br_ + img * (H_ * WFP);
  const _Float16* Bsi = Bi_ + img * (H_ * WFP);

  auto stage = [&](int kk, int buf) {
    for (int idx = tid; idx < 4096; idx += 96) {  // transpose-stage 32x128
      int h = idx >> 7, w = idx & 127, hg = kk + h;
      _Float16 vr = (_Float16)0.0f, vi = (_Float16)0.0f;
      if (hg < H_) { vr = Bsr[hg * WFP + w]; vi = Bsi[hg * WFP + w]; }
      BT[0][buf][w * 32 + h] = vr;
      BT[1][buf][w * 32 + h] = vi;
    }
  };
  auto prefetch_slab = [&](int kk) {  // 32 rows x 256B x 2 planes = 128 lines
    for (int idx = tid; idx < 128; idx += 96) {
      int plane = idx >> 6, h = (idx & 63) >> 1, part = idx & 1;
      if (kk + h < H_) {
        const _Float16* p =
            (plane ? Bsi : Bsr) + (kk + h) * WFP + part * 64;
        __builtin_prefetch(p, 0, 0);
      }
    }
  };

  v8f zr[8], zi[8];
#pragma unroll
  for (int t = 0; t < 8; ++t) { zr[t] = zero8f(); zi[t] = zero8f(); }

  stage(0, 0);
  __syncthreads();
  int buf = 0;
  for (int ks = 0; ks < 8; ++ks) {
    int kk = ks * 32;
    if (ks < 7) prefetch_slab(kk + 32);  // warm cache while we compute
    v16h ap = frag_a(P,  256, ra, kk, half);
    v16h a1 = frag_a(A1, 256, ra, kk, half);
    v16h a2 = frag_a(A2, 256, ra, kk, half);
#pragma unroll
    for (int t = 0; t < 8; ++t) {
      int col = t * 16 + l15;
      v16h br = frag_b(&BT[0][buf][0], 32, col, 0, half);
      v16h bi = frag_b(&BT[1][buf][0], 32, col, 0, half);
      zr[t] = wmma(ap, br, zr[t]);
      zr[t] = wmma(a1, bi, zr[t]);
      zi[t] = wmma(ap, bi, zi[t]);
      zi[t] = wmma(a2, br, zi[t]);
    }
    if (ks < 7) stage(kk + 32, buf ^ 1);  // fill other buffer (likely cached)
    __syncthreads();
    buf ^= 1;
  }
#pragma unroll
  for (int t = 0; t < 8; ++t) {
    int col = t * 16 + l15;
#pragma unroll
    for (int v = 0; v < 8; ++v) {
      int row = rw + v + 8 * half;
      Or[(img * H_ + row) * WFP + col] = (_Float16)zr[t][v];
      Oi[(img * H_ + row) * WFP + col] = (_Float16)zi[t][v];
    }
  }
}

// ----------- stage 3: block-diagonal complex mixing (2 layers) -------------
// grid (908, 8); 128 threads = 4 waves; per block M=64 rows of (b,h,w).
__global__ __launch_bounds__(128) void mix_kernel(
    const _Float16* __restrict__ Xr, const _Float16* __restrict__ Xi,
    const float* __restrict__ w1, const float* __restrict__ b1,
    const float* __restrict__ w2, const float* __restrict__ b2,
    _Float16* __restrict__ Or, _Float16* __restrict__ Oi) {
  extern __shared__ char smem[];
  _Float16* Wt1r = (_Float16*)smem;          // [o][i] 96x96
  _Float16* Wt1i = Wt1r + BS * BS;
  _Float16* Wt2r = Wt1i + BS * BS;
  _Float16* Wt2i = Wt2r + BS * BS;
  _Float16* A1r  = Wt2i + BS * BS;           // [64][96]
  _Float16* A1i  = A1r + 64 * BS;
  _Float16* Actr = A1i + 64 * BS;
  _Float16* Acti = Actr + 64 * BS;
  float* bias = (float*)(Acti + 64 * BS);    // b1r|b1i|b2r|b2i, 96 each
  int tid = threadIdx.x;
  int wave = tid >> 5, lane = tid & 31, half = lane >> 4, l15 = lane & 15;
  int n = blockIdx.y;
  int m0 = blockIdx.x * 64;

  for (int idx = tid; idx < BS * BS; idx += 128) {
    int i = idx / BS, o = idx % BS;
    int g = ((n * BS + i) * BS + o) * 2;
    Wt1r[o * BS + i] = (_Float16)w1[g];
    Wt1i[o * BS + i] = (_Float16)w1[g + 1];
    Wt2r[o * BS + i] = (_Float16)w2[g];
    Wt2i[o * BS + i] = (_Float16)w2[g + 1];
  }
  if (tid < BS) {
    bias[tid]           = b1[(n * BS + tid) * 2];
    bias[96 + tid]      = b1[(n * BS + tid) * 2 + 1];
    bias[192 + tid]     = b2[(n * BS + tid) * 2];
    bias[288 + tid]     = b2[(n * BS + tid) * 2 + 1];
  }
  // gather A tile; l fastest over threads -> consecutive w (coalesced f16)
  for (int idx = tid; idx < 64 * BS; idx += 128) {
    int l = idx & 63, i = idx >> 6;
    int m = m0 + l;
    _Float16 vr = (_Float16)0.0f, vi = (_Float16)0.0f;
    if (m < MIXM) {
      int b = m / (H_ * WF), rem = m % (H_ * WF);
      int h = rem / WF, w = rem % WF;
      int g = ((b * C_ + n * BS + i) * H_ + h) * WFP + w;
      vr = Xr[g]; vi = Xi[g];
    }
    A1r[l * BS + i] = vr; A1i[l * BS + i] = vi;
  }
  __syncthreads();

  int lw = wave * 16;
  v8f yr[6], yi[6];
#pragma unroll
  for (int t = 0; t < 6; ++t) { yr[t] = zero8f(); yi[t] = zero8f(); }
#pragma unroll
  for (int ks = 0; ks < 3; ++ks) {  // layer 1, K=96
    int kk = ks * 32;
    v16h ar = frag_a(A1r, BS, lw + l15, kk, half);
    v16h ai = frag_a(A1i, BS, lw + l15, kk, half);
    v16h an = neg16(ai);
#pragma unroll
    for (int t = 0; t < 6; ++t) {
      int col = t * 16 + l15;
      v16h wr_ = frag_b(Wt1r, BS, col, kk, half);
      v16h wi_ = frag_b(Wt1i, BS, col, kk, half);
      yr[t] = wmma(ar, wr_, yr[t]);
      yr[t] = wmma(an, wi_, yr[t]);
      yi[t] = wmma(ar, wi_, yi[t]);
      yi[t] = wmma(ai, wr_, yi[t]);
    }
  }
#pragma unroll
  for (int t = 0; t < 6; ++t) {  // bias + sparsity mask -> Act
    int col = t * 16 + l15;
    float br_ = bias[col], bi_ = bias[96 + col];
#pragma unroll
    for (int v = 0; v < 8; ++v) {
      float xr = yr[t][v] + br_, xi = yi[t][v] + bi_;
      float mag = sqrtf(xr * xr + xi * xi + 1e-8f);
      if (mag <= 0.01f) { xr = 0.0f; xi = 0.0f; }
      int rl = lw + v + 8 * half;
      Actr[rl * BS + col] = (_Float16)xr;
      Acti[rl * BS + col] = (_Float16)xi;
    }
  }
  __syncthreads();

  v8f zr[6], zi[6];
#pragma unroll
  for (int t = 0; t < 6; ++t) { zr[t] = zero8f(); zi[t] = zero8f(); }
#pragma unroll
  for (int ks = 0; ks < 3; ++ks) {  // layer 2
    int kk = ks * 32;
    v16h ar = frag_a(Actr, BS, lw + l15, kk, half);
    v16h ai = frag_a(Acti, BS, lw + l15, kk, half);
    v16h an = neg16(ai);
#pragma unroll
    for (int t = 0; t < 6; ++t) {
      int col = t * 16 + l15;
      v16h wr_ = frag_b(Wt2r, BS, col, kk, half);
      v16h wi_ = frag_b(Wt2i, BS, col, kk, half);
      zr[t] = wmma(ar, wr_, zr[t]);
      zr[t] = wmma(an, wi_, zr[t]);
      zi[t] = wmma(ar, wi_, zi[t]);
      zi[t] = wmma(ai, wr_, zi[t]);
    }
  }
#pragma unroll
  for (int t = 0; t < 6; ++t) {  // bias2 + scatter-store
    int col = t * 16 + l15;
    float br2 = bias[192 + col], bi2 = bias[288 + col];
#pragma unroll
    for (int v = 0; v < 8; ++v) {
      int m = m0 + lw + v + 8 * half;
      if (m < MIXM) {
        int b = m / (H_ * WF), rem = m % (H_ * WF);
        int h = rem / WF, w = rem % WF;
        int g = ((b * C_ + n * BS + col) * H_ + h) * WFP + w;
        Or[g] = (_Float16)(zr[t][v] + br2);
        Oi[g] = (_Float16)(zi[t][v] + bi2);
      }
    }
  }
}

// ------------- stage 5: irfft along W (real-output GEMM) -------------------
// out = Zr*IwcT + Zi*IwsT ; M=368640 (64/block), N=240 (15 tiles), K=128.
__global__ __launch_bounds__(128) void irfft_w_kernel(
    const _Float16* __restrict__ Zr, const _Float16* __restrict__ Zi,
    const _Float16* __restrict__ IcT, const _Float16* __restrict__ IsT,
    float* __restrict__ out) {
  int wave = threadIdx.x >> 5, lane = threadIdx.x & 31;
  int half = lane >> 4, l15 = lane & 15;
  int rw = blockIdx.x * 64 + wave * 16;
  int ra = rw + l15;
  v8f acc[15];
#pragma unroll
  for (int t = 0; t < 15; ++t) acc[t] = zero8f();
#pragma unroll
  for (int ks = 0; ks < 4; ++ks) {
    int kk = ks * 32;
    if (kk + 32 < WFP) __builtin_prefetch(Zr + ra * WFP + kk + 32, 0, 0);
    v16h ar = frag_a(Zr, WFP, ra, kk, half);
    v16h ai = frag_a(Zi, WFP, ra, kk, half);
    v16h bc = frag_b(IcT, WFP, l15, kk, half);
    v16h bs = frag_b(IsT, WFP, l15, kk, half);
#pragma unroll
    for (int t = 0; t < 15; ++t) {
      v16h bcn = bc, bsn = bs;
      if (t < 14) {
        int coln = (t + 1) * 16 + l15;
        bcn = frag_b(IcT, WFP, coln, kk, half);
        bsn = frag_b(IsT, WFP, coln, kk, half);
      }
      acc[t] = wmma(ar, bc, acc[t]);
      acc[t] = wmma(ai, bs, acc[t]);
      bc = bcn; bs = bsn;
    }
  }
#pragma unroll
  for (int t = 0; t < 15; ++t) {
    int col = t * 16 + l15;
#pragma unroll
    for (int v = 0; v < 8; ++v) out[(rw + v + 8 * half) * W_ + col] = acc[t][v];
  }
}

// ------------------------------- launch ------------------------------------
extern "C" void kernel_launch(void* const* d_in, const int* in_sizes, int n_in,
                              void* d_out, int out_size, void* d_ws,
                              size_t ws_size, hipStream_t stream) {
  (void)in_sizes; (void)n_in; (void)out_size; (void)ws_size;
  const float* x  = (const float*)d_in[0];
  const float* w1 = (const float*)d_in[1];
  const float* b1 = (const float*)d_in[2];
  const float* w2 = (const float*)d_in[3];
  const float* b2 = (const float*)d_in[4];
  float* out = (float*)d_out;

  // ws: one f16 plane-pair (S2r/S2i) + twiddles (~181 MB). One more plane-pair lives
  // in the front of d_out (fully overwritten by the final kernel).
  _Float16* S2r  = (_Float16*)d_ws;
  _Float16* S2i  = S2r + PLANE;
  _Float16* FwcT = S2i + PLANE;
  _Float16* FwsT = FwcT + 128 * 256;
  _Float16* Pm   = FwsT + 128 * 256;
  _Float16* Sp   = Pm + 240 * 256;     // +sin
  _Float16* Sn   = Sp + 240 * 256;     // -sin
  _Float16* IwcT = Sn + 240 * 256;
  _Float16* IwsT = IwcT + 240 * 128;
  _Float16* S1r  = (_Float16*)d_out;   // d_out-carved scratch planes
  _Float16* S1i  = S1r + PLANE;

  init_fw<<<128, 256, 0, stream>>>(FwcT, FwsT);
  init_h <<<240, 256, 0, stream>>>(Pm, Sp, Sn);
  init_iw<<<120, 256, 0, stream>>>(IwcT, IwsT);

  // 1) rfft along W: x -> S1 (in d_out space)
  rfft_w_kernel<<<ROWS_W / 64, 128, 0, stream>>>(x, FwcT, FwsT, S1r, S1i);
  // 2) forward fft along H: S1 -> S2   (Zr=P*Br+S*Bi, Zi=P*Bi-S*Br)
  cgemm_h_kernel<<<dim3(5, IMGS), 96, 0, stream>>>(Pm, Sp, Sn, S1r, S1i,
                                                   S2r, S2i);
  // 3) channel mixing: S2 -> S3 (reuse S1 space)
  mix_kernel<<<dim3((MIXM + 63) / 64, NB), 128, 124416, stream>>>(
      S2r, S2i, w1, b1, w2, b2, S1r, S1i);
  // 4) inverse fft along H: S3 -> S4 (reuse S2 space); swap sin matrices
  cgemm_h_kernel<<<dim3(5, IMGS), 96, 0, stream>>>(Pm, Sn, Sp, S1r, S1i,
                                                   S2r, S2i);
  // 5) irfft along W (hermitian fold): S4 -> out (overwrites scratch planes)
  irfft_w_kernel<<<ROWS_W / 64, 128, 0, stream>>>(S2r, S2i, IwcT, IwsT, out);
}
